// RSNN_74440373174661
// MI455X (gfx1250) — matrix operand order repaired and checked by reading, not compiled, not run on hardware
//
#include <hip/hip_runtime.h>
#include <hip/hip_bf16.h>

// ---------------------------------------------------------------------------
// RSNN on MI455X (gfx1250, wave32): fp16 WMMA (f32 accumulate) for the
// recurrent / input / readout GEMMs; sequential time loop driven from host.
// ---------------------------------------------------------------------------

typedef __attribute__((ext_vector_type(16))) _Float16 v16h;
typedef __attribute__((ext_vector_type(8)))  _Float16 v8h;
typedef __attribute__((ext_vector_type(8)))  float    v8f;

#define NTRIALS 256
#define TSTEPS  200
#define IDIM    128
#define NDIM    1024
#define ODIM    128

__device__ __forceinline__ v8f wmma16x16x32(v16h a, v16h b, v8f c) {
    // D = A(16x32 f16) * B(32x16 f16) + C(16x16 f32)
    return __builtin_amdgcn_wmma_f32_16x16x32_f16(
        /*neg_a=*/false, a, /*neg_b=*/false, b,
        /*c_mod=*/(short)0, c, /*reuse_a=*/false, /*reuse_b=*/false);
}

__device__ __forceinline__ float sigmoidf(float x) {
    return 1.0f / (1.0f + __expf(-x));
}

// A-fragment (16x32, f16) from a row-major f16 matrix, row stride ld.
// Lane L: M = L%16; halves 0..7 = K[k0 + (L/16)*8 ..], halves 8..15 = K[k0+16+(L/16)*8 ..]
__device__ __forceinline__ v16h load_a_f16(const _Float16* __restrict__ p,
                                           int row, int ld, int k0,
                                           int m16, int hi) {
    const _Float16* ap = p + (size_t)(row + m16) * ld + k0 + hi * 8;
    v8h lo = *reinterpret_cast<const v8h*>(ap);
    v8h hh = *reinterpret_cast<const v8h*>(ap + 16);
    v16h a;
#pragma unroll
    for (int e = 0; e < 8; ++e) { a[e] = lo[e]; a[8 + e] = hh[e]; }
    return a;
}

// A-fragment from row-major f32 data (converted inline to f16).
__device__ __forceinline__ v16h load_a_f32(const float* __restrict__ p,
                                           size_t rowOff, int k0,
                                           int hi) {
    const float* ap = p + rowOff + k0 + hi * 8;
    v8f lo = *reinterpret_cast<const v8f*>(ap);
    v8f hh = *reinterpret_cast<const v8f*>(ap + 16);
    v16h a;
#pragma unroll
    for (int e = 0; e < 8; ++e) {
        a[e]     = (_Float16)lo[e];
        a[8 + e] = (_Float16)hh[e];
    }
    return a;
}

// B-fragment (32x16, f16) from weight matrix stored row-major [n][k], stride ld.
// Lane L: N = L%16; halves 0..7 = K[k0 + (L/16)*8 ..], halves 8..15 = K[k0+16+(L/16)*8 ..]
__device__ __forceinline__ v16h load_b_f16(const _Float16* __restrict__ p,
                                           int col, int ld, int k0,
                                           int m16, int hi) {
    const _Float16* bp = p + (size_t)(col + m16) * ld + k0 + hi * 8;
    v8h lo = *reinterpret_cast<const v8h*>(bp);
    v8h hh = *reinterpret_cast<const v8h*>(bp + 16);
    v16h b;
#pragma unroll
    for (int e = 0; e < 8; ++e) { b[e] = lo[e]; b[8 + e] = hh[e]; }
    return b;
}

// --------------------------- one-time kernels ------------------------------

__global__ void cvt_f32_to_f16(const float* __restrict__ src,
                               _Float16* __restrict__ dst, int n) {
    int i = blockIdx.x * blockDim.x + threadIdx.x;
    if (i < n) dst[i] = (_Float16)src[i];
}

// v = 0, s16 = sigmoid(0) = 0.5, vt[:,0,:] = 0, zt[:,0,:] = 0
__global__ void init_state(float* __restrict__ v, _Float16* __restrict__ s0,
                           float* __restrict__ vt, float* __restrict__ zt) {
    int i = blockIdx.x * blockDim.x + threadIdx.x;
    if (i < NTRIALS * NDIM) {
        v[i]  = 0.0f;
        s0[i] = (_Float16)0.5f;
        int b = i >> 10;          // / NDIM
        int c = i & (NDIM - 1);   // % NDIM
        size_t idx = (size_t)b * (size_t)(TSTEPS + 1) * NDIM + c; // t = 0 row
        vt[idx] = 0.0f;
        zt[idx] = 0.0f;
    }
}

// --------------------------- per-step kernels ------------------------------

// acc[b,n] = sum_k sigmoid(v)[b,k]*J[n,k] + sum_i inp[b,t,i]*B[n,i]
// v_new = 0.9*v + 0.1*acc;  spike, vt/zt/outs bookkeeping; s16/tanh16 for next.
// Block: 128 threads (4 waves, 2x2 of 32x32), tile 64(M) x 64(N).
__global__ void rsnn_step(const _Float16* __restrict__ sCur,
                          const float*    __restrict__ inputs,
                          const _Float16* __restrict__ J16,
                          const _Float16* __restrict__ B16,
                          const float*    __restrict__ noise,
                          float*          __restrict__ v,
                          float*          __restrict__ vt,
                          float*          __restrict__ zt,
                          _Float16*       __restrict__ sNext,
                          _Float16*       __restrict__ th16,
                          int t) {
    const int lane = threadIdx.x & 31;
    const int wave = threadIdx.x >> 5;
    const int wm = wave >> 1, wn = wave & 1;
    const int rowBase = blockIdx.y * 64 + wm * 32;   // batch tile
    const int colBase = blockIdx.x * 64 + wn * 32;   // neuron tile
    const int m16 = lane & 15;
    const int hi  = lane >> 4;

    v8f c[2][2] = {};

    // ---- recurrent term: sigmoid(v) (f16, ld=NDIM)  x  J16 [n][k] ----
#pragma unroll 4
    for (int k0 = 0; k0 < NDIM; k0 += 32) {
        v16h a0 = load_a_f16(sCur, rowBase,      NDIM, k0, m16, hi);
        v16h a1 = load_a_f16(sCur, rowBase + 16, NDIM, k0, m16, hi);
        v16h b0 = load_b_f16(J16,  colBase,      NDIM, k0, m16, hi);
        v16h b1 = load_b_f16(J16,  colBase + 16, NDIM, k0, m16, hi);
        c[0][0] = wmma16x16x32(a0, b0, c[0][0]);
        c[0][1] = wmma16x16x32(a0, b1, c[0][1]);
        c[1][0] = wmma16x16x32(a1, b0, c[1][0]);
        c[1][1] = wmma16x16x32(a1, b1, c[1][1]);
    }

    // ---- input term: inputs[b,t,:] (f32->f16 inline)  x  B16 [n][i] ----
#pragma unroll
    for (int k0 = 0; k0 < IDIM; k0 += 32) {
        size_t r0 = ((size_t)(rowBase + m16)      * TSTEPS + t) * IDIM;
        size_t r1 = ((size_t)(rowBase + 16 + m16) * TSTEPS + t) * IDIM;
        v16h a0 = load_a_f32(inputs, r0, k0, hi);
        v16h a1 = load_a_f32(inputs, r1, k0, hi);
        v16h b0 = load_b_f16(B16, colBase,      IDIM, k0, m16, hi);
        v16h b1 = load_b_f16(B16, colBase + 16, IDIM, k0, m16, hi);
        c[0][0] = wmma16x16x32(a0, b0, c[0][0]);
        c[0][1] = wmma16x16x32(a0, b1, c[0][1]);
        c[1][0] = wmma16x16x32(a1, b0, c[1][0]);
        c[1][1] = wmma16x16x32(a1, b1, c[1][1]);
    }

    // ---- epilogue: leaky integration + spike + state for next step ----
#pragma unroll
    for (int i = 0; i < 2; ++i) {
#pragma unroll
        for (int j = 0; j < 2; ++j) {
#pragma unroll
            for (int r = 0; r < 8; ++r) {
                int row = rowBase + i * 16 + hi * 8 + r;   // batch index
                int col = colBase + j * 16 + m16;          // neuron index
                size_t sIdx = (size_t)row * NDIM + col;

                float acc  = c[i][j][r];
                float vold = v[sIdx];
                float vn   = 0.9f * vold + 0.1f * acc;     // (1-a)v + a*acc
                v[sIdx] = vn;

                size_t seqIdx = (size_t)row * (size_t)(TSTEPS + 1) * NDIM
                              + (size_t)(t + 1) * NDIM + col;
                vt[seqIdx] = vn;

                // spike: sigmoid(TEMP*(vn-THR)/THR) > u ; TEMP=0.1, THR=0.4
                float u = noise[((size_t)row * TSTEPS + t) * NDIM + col];
                float p = sigmoidf(0.25f * (vn - 0.4f));   // 0.1/0.4 = 0.25
                zt[seqIdx] = (p > u) ? 1.0f : 0.0f;

                sNext[sIdx] = (_Float16)sigmoidf(vn);
                th16[sIdx]  = (_Float16)tanhf(vn);
            }
        }
    }
}

// outs[b,t,o] = sum_k tanh(v_new)[b,k] * W[o,k]
__global__ void rsnn_readout(const _Float16* __restrict__ th16,
                             const _Float16* __restrict__ W16,
                             float*          __restrict__ outs,
                             int t) {
    const int lane = threadIdx.x & 31;
    const int wave = threadIdx.x >> 5;
    const int wm = wave >> 1, wn = wave & 1;
    const int rowBase = blockIdx.y * 64 + wm * 32;   // batch
    const int colBase = blockIdx.x * 64 + wn * 32;   // output dim
    const int m16 = lane & 15;
    const int hi  = lane >> 4;

    v8f c[2][2] = {};
#pragma unroll 4
    for (int k0 = 0; k0 < NDIM; k0 += 32) {
        v16h a0 = load_a_f16(th16, rowBase,      NDIM, k0, m16, hi);
        v16h a1 = load_a_f16(th16, rowBase + 16, NDIM, k0, m16, hi);
        v16h b0 = load_b_f16(W16,  colBase,      NDIM, k0, m16, hi);
        v16h b1 = load_b_f16(W16,  colBase + 16, NDIM, k0, m16, hi);
        c[0][0] = wmma16x16x32(a0, b0, c[0][0]);
        c[0][1] = wmma16x16x32(a0, b1, c[0][1]);
        c[1][0] = wmma16x16x32(a1, b0, c[1][0]);
        c[1][1] = wmma16x16x32(a1, b1, c[1][1]);
    }
#pragma unroll
    for (int i = 0; i < 2; ++i)
#pragma unroll
        for (int j = 0; j < 2; ++j)
#pragma unroll
            for (int r = 0; r < 8; ++r) {
                int row = rowBase + i * 16 + hi * 8 + r;
                int col = colBase + j * 16 + m16;
                outs[((size_t)row * TSTEPS + t) * ODIM + col] = c[i][j][r];
            }
}

// ------------------------------- launcher ----------------------------------

extern "C" void kernel_launch(void* const* d_in, const int* in_sizes, int n_in,
                              void* d_out, int out_size, void* d_ws, size_t ws_size,
                              hipStream_t stream) {
    (void)in_sizes; (void)n_in; (void)out_size; (void)ws_size;

    const float* inputs = (const float*)d_in[0];  // (256,200,128)
    const float* noise  = (const float*)d_in[1];  // (256,200,1024)
    const float* J      = (const float*)d_in[2];  // (1024,1024)
    const float* Bm     = (const float*)d_in[3];  // (1024,128)
    const float* W      = (const float*)d_in[4];  // (128,1024)

    // outputs, flat: vt (256,201,1024) | zt (256,201,1024) | outs (256,200,128)
    float* vt   = (float*)d_out;
    float* zt   = vt + (size_t)NTRIALS * (TSTEPS + 1) * NDIM;
    float* outs = zt + (size_t)NTRIALS * (TSTEPS + 1) * NDIM;

    // workspace layout (all offsets 256B-aligned)
    char* ws = (char*)d_ws;
    float*    v    = (float*)ws;      ws += (size_t)NTRIALS * NDIM * sizeof(float);
    _Float16* s16a = (_Float16*)ws;   ws += (size_t)NTRIALS * NDIM * sizeof(_Float16);
    _Float16* s16b = (_Float16*)ws;   ws += (size_t)NTRIALS * NDIM * sizeof(_Float16);
    _Float16* th16 = (_Float16*)ws;   ws += (size_t)NTRIALS * NDIM * sizeof(_Float16);
    _Float16* J16  = (_Float16*)ws;   ws += (size_t)NDIM * NDIM * sizeof(_Float16);
    _Float16* B16  = (_Float16*)ws;   ws += (size_t)NDIM * IDIM * sizeof(_Float16);
    _Float16* W16  = (_Float16*)ws;   ws += (size_t)ODIM * NDIM * sizeof(_Float16);

    // one-time: weight conversion + state init
    {
        int n = NDIM * NDIM;
        cvt_f32_to_f16<<<(n + 255) / 256, 256, 0, stream>>>(J, J16, n);
        n = NDIM * IDIM;
        cvt_f32_to_f16<<<(n + 255) / 256, 256, 0, stream>>>(Bm, B16, n);
        n = ODIM * NDIM;
        cvt_f32_to_f16<<<(n + 255) / 256, 256, 0, stream>>>(W, W16, n);
        n = NTRIALS * NDIM;
        init_state<<<(n + 255) / 256, 256, 0, stream>>>(v, s16a, vt, zt);
    }

    dim3 gridG(NDIM / 64, NTRIALS / 64);   // (16, 4)
    dim3 gridO(ODIM / 64, NTRIALS / 64);   // (2, 4)

    for (int t = 0; t < TSTEPS; ++t) {
        const _Float16* sCur  = (t & 1) ? s16b : s16a;
        _Float16*       sNext = (t & 1) ? s16a : s16b;
        rsnn_step<<<gridG, 128, 0, stream>>>(sCur, inputs, J16, B16, noise,
                                             v, vt, zt, sNext, th16, t);
        rsnn_readout<<<gridO, 128, 0, stream>>>(th16, W16, outs, t);
    }
}
